// TransformerEncoder_34119220199497
// MI455X (gfx1250) — compile-verified, matrix operand order checked by reading, not run on hardware
//
#include <hip/hip_runtime.h>
#include <hip/hip_bf16.h>

// ---------------------------------------------------------------------------
// Transformer encoder forward (B=2, S=2048, E=512, H=8, D=64, L=2, DFF=2048)
// fp32 reference; matmuls run as bf16 WMMA (v_wmma_f32_16x16x32_bf16) with
// fp32 accumulation. Flash-attention style online softmax (no SxS matrix).
// ---------------------------------------------------------------------------

#define BDIM 2
#define SEQ  2048
#define EMB  512
#define NH   8
#define HD   64
#define NL   2
#define DFF_ 2048
#define ROWS (BDIM * SEQ)

typedef __attribute__((ext_vector_type(16))) __bf16          v16bf;
typedef __attribute__((ext_vector_type(8)))  float           v8f;
typedef __attribute__((ext_vector_type(8)))  unsigned short  ush8;

__device__ __forceinline__ unsigned short f2bf(float f) {
    unsigned int u = __float_as_uint(f);
    unsigned int r = (u + 0x7FFFu + ((u >> 16) & 1u)) >> 16;   // RNE
    return (unsigned short)r;
}

// Load one 16x32 bf16 fragment (A layout; identical code serves B fragments of
// A@W^T because a row of row-major W[N,K] is a column of B).
// Per CDNA5 ISA: lanes 0-15 hold K k0..k0+7 (elems 0..7) and
// k0+16..k0+23 (elems 8..15); lanes 16-31 are offset by 8 in K.
__device__ __forceinline__ v16bf frag_ld(const unsigned short* base, int ld,
                                         int row, int k0, int lane) {
    int h = (lane >> 4) & 1;
    const unsigned short* p = base + (size_t)(row + (lane & 15)) * ld + k0 + h * 8;
    union { v16bf v; ush8 u[2]; } r;
    r.u[0] = *(const ush8*)p;
    r.u[1] = *(const ush8*)(p + 16);
    return r.v;
}

__device__ __forceinline__ v8f wmma_bf16(v16bf a, v16bf b, v8f c) {
    return __builtin_amdgcn_wmma_f32_16x16x32_bf16(false, a, false, b,
                                                   (short)0, c, false, false);
}

// ---------------------------------------------------------------------------
// fp32 -> bf16 conversion (weights)
// ---------------------------------------------------------------------------
__global__ void k_f32_to_bf16(const float* __restrict__ in,
                              unsigned short* __restrict__ out, int n) {
    int i = blockIdx.x * 256 + threadIdx.x;
    if (i < n) out[i] = f2bf(in[i]);
}

// ---------------------------------------------------------------------------
// Row LayerNorm over E=512. One row per 256-thread block.
// y = LN(in)[*g + b]; t = y + residual (opt); write fp32 and/or bf16.
// ---------------------------------------------------------------------------
__global__ __launch_bounds__(256)
void k_layernorm(const float* __restrict__ in, const float* g, const float* b,
                 const float* res, float* outF, unsigned short* outB) {
    __shared__ float s1[256], s2[256];
    int r = blockIdx.x, t = threadIdx.x;
    const float* row = in + (size_t)r * EMB;
    float a0 = row[t], a1 = row[t + 256];
    s1[t] = a0 + a1;
    s2[t] = a0 * a0 + a1 * a1;
    __syncthreads();
    for (int o = 128; o > 0; o >>= 1) {
        if (t < o) { s1[t] += s1[t + o]; s2[t] += s2[t + o]; }
        __syncthreads();
    }
    float mu  = s1[0] * (1.0f / EMB);
    float var = s2[0] * (1.0f / EMB) - mu * mu;
    float rs  = rsqrtf(var + 1e-5f);
    #pragma unroll
    for (int e = 0; e < 2; e++) {
        int   c = t + e * 256;
        float v = ((e ? a1 : a0) - mu) * rs;
        if (g) v = v * g[c] + b[c];
        if (res) v += res[(size_t)r * EMB + c];
        if (outF) outF[(size_t)r * EMB + c] = v;
        if (outB) outB[(size_t)r * EMB + c] = f2bf(v);
    }
}

// ---------------------------------------------------------------------------
// C[M,N] = A[M,K](bf16) @ W[N,K]^T(bf16) + bias [+ReLU] [+residual]
// Block: 256 threads = 8 waves; tile 256(M) x 64(N).
// Each wave: 2 M-subtiles x 4 N-subtiles = 8 WMMAs per staged 32-wide k-step.
// B fragments preloaded into distinct registers so the WMMA chain is not
// fenced per-fragment by s_wait_dscnt 0.
// ---------------------------------------------------------------------------
__global__ __launch_bounds__(256)
void k_gemm_bf16(const unsigned short* __restrict__ A,
                 const unsigned short* __restrict__ W,
                 const float* __restrict__ bias, const float* __restrict__ res,
                 float* outF, unsigned short* outB,
                 int M, int N, int K, int relu) {
    __shared__ unsigned short Ws[64 * 40];            // 64 N-rows x 32 K (pad 40)
    int lane = threadIdx.x & 31;
    int wave = threadIdx.x >> 5;                       // 0..7 -> M sub-tile pair
    int bm = blockIdx.y * 256;
    int bn = blockIdx.x * 64;
    int rowA = bm + wave * 32;

    v8f zero = {0.f, 0.f, 0.f, 0.f, 0.f, 0.f, 0.f, 0.f};
    v8f acc[2][4];
    #pragma unroll
    for (int r = 0; r < 2; r++)
        #pragma unroll
        for (int j = 0; j < 4; j++) acc[r][j] = zero;

    for (int k0 = 0; k0 < K; k0 += 32) {
        __syncthreads();
        {   // cooperative stage of W tile: rows bn..bn+63, cols k0..k0+31
            int idx = threadIdx.x * 8;
            int rr = idx >> 5, cc = idx & 31;
            ush8 v = *(const ush8*)(W + (size_t)(bn + rr) * K + k0 + cc);
            *(ush8*)(Ws + rr * 40 + cc) = v;
            if (k0 + 32 < K)   // global_prefetch_b8 of next W tile
                __builtin_prefetch((const void*)(W + (size_t)(bn + rr) * K + k0 + 32 + cc), 0, 3);
        }
        __syncthreads();
        v16bf a0 = frag_ld(A, K, rowA,      k0, lane);
        v16bf a1 = frag_ld(A, K, rowA + 16, k0, lane);
        v16bf bfr[4];
        #pragma unroll
        for (int j = 0; j < 4; j++) bfr[j] = frag_ld(Ws, 40, j * 16, 0, lane);
        #pragma unroll
        for (int j = 0; j < 4; j++) acc[0][j] = wmma_bf16(a0, bfr[j], acc[0][j]);
        #pragma unroll
        for (int j = 0; j < 4; j++) acc[1][j] = wmma_bf16(a1, bfr[j], acc[1][j]);
    }

    int h = (lane >> 4) & 1, c16 = lane & 15;
    #pragma unroll
    for (int r = 0; r < 2; r++) {
        #pragma unroll
        for (int j = 0; j < 4; j++) {
            int col = bn + j * 16 + c16;
            float bv = bias ? bias[col] : 0.f;
            #pragma unroll
            for (int i = 0; i < 8; i++) {
                int rowc = bm + wave * 32 + r * 16 + i + h * 8;
                float v = acc[r][j][i] + bv;
                if (relu) v = fmaxf(v, 0.f);
                if (res)  v += res[(size_t)rowc * N + col];
                if (outF) outF[(size_t)rowc * N + col] = v;
                if (outB) outB[(size_t)rowc * N + col] = f2bf(v);
            }
        }
    }
}

// ---------------------------------------------------------------------------
// Flash attention. Block = (qtile=64 rows, head, batch), 128 threads = 4 waves,
// each wave owns 16 query rows. Streams 32-key tiles through LDS (K row-major,
// V transposed), online softmax, PV accumulated in fp32 via WMMA.
// ---------------------------------------------------------------------------
__global__ __launch_bounds__(128)
void k_attention(const unsigned short* __restrict__ Q,
                 const unsigned short* __restrict__ Kb,
                 const unsigned short* __restrict__ V,
                 const unsigned char* __restrict__ mask,
                 float* __restrict__ out) {
    __shared__ unsigned short Ks[32 * 72];     // 32 keys x 64 d   (pad 72)
    __shared__ unsigned short Vt[64 * 40];     // 64 d    x 32 key (pad 40)
    __shared__ unsigned short Pb[4 * 16 * 40]; // per-wave P tile 16 x 32
    __shared__ float mb[32];

    int lane = threadIdx.x & 31;
    int wave = threadIdx.x >> 5;               // 0..3
    int q0 = blockIdx.x * 64;
    int hh = blockIdx.y;
    int bb = blockIdx.z;
    const size_t baseRow = (size_t)bb * SEQ;
    const unsigned short* Qb    = Q  + baseRow * EMB + hh * HD;
    const unsigned short* Kbase = Kb + baseRow * EMB + hh * HD;
    const unsigned short* Vbase = V  + baseRow * EMB + hh * HD;

    v16bf qf0 = frag_ld(Qb, EMB, q0 + wave * 16, 0,  lane);
    v16bf qf1 = frag_ld(Qb, EMB, q0 + wave * 16, 32, lane);

    v8f zero = {0.f, 0.f, 0.f, 0.f, 0.f, 0.f, 0.f, 0.f};
    v8f acc[4];
    #pragma unroll
    for (int j = 0; j < 4; j++) acc[j] = zero;
    float mrow[8], lrow[8];
    #pragma unroll
    for (int i = 0; i < 8; i++) { mrow[i] = -1e30f; lrow[i] = 0.f; }

    unsigned short* Pw = Pb + wave * 16 * 40;
    const float scale = 0.125f;                // 1/sqrt(64)
    int c16 = lane & 15, hf = (lane >> 4) & 1;

    for (int key0 = 0; key0 < SEQ; key0 += 32) {
        __syncthreads();
        {   // stage K tile (row-major) and V^T tile
            int idx = threadIdx.x * 16;
            int rr = idx >> 6, cc = idx & 63;
            ush8 v0 = *(const ush8*)(Kbase + (size_t)(key0 + rr) * EMB + cc);
            ush8 v1 = *(const ush8*)(Kbase + (size_t)(key0 + rr) * EMB + cc + 8);
            *(ush8*)(Ks + rr * 72 + cc)     = v0;
            *(ush8*)(Ks + rr * 72 + cc + 8) = v1;
            #pragma unroll
            for (int e = 0; e < 16; e++) {
                int i2 = threadIdx.x * 16 + e;
                int kk = i2 >> 6, dd = i2 & 63;
                Vt[dd * 40 + kk] = Vbase[(size_t)(key0 + kk) * EMB + dd];
            }
            if (threadIdx.x < 32)
                mb[threadIdx.x] = mask[bb * SEQ + key0 + threadIdx.x] ? -1e30f : 0.f;
        }
        __syncthreads();

        // logits 16(q) x 32(key): preload all 4 K fragments, then 4 WMMAs
        v16bf kf[4];
        kf[0] = frag_ld(Ks, 72, 0,  0,  lane);
        kf[1] = frag_ld(Ks, 72, 0,  32, lane);
        kf[2] = frag_ld(Ks, 72, 16, 0,  lane);
        kf[3] = frag_ld(Ks, 72, 16, 32, lane);
        v8f s0 = zero, s1 = zero;
        s0 = wmma_bf16(qf0, kf[0], s0);
        s0 = wmma_bf16(qf1, kf[1], s0);
        s1 = wmma_bf16(qf0, kf[2], s1);
        s1 = wmma_bf16(qf1, kf[3], s1);

        float bias0 = mb[c16], bias1 = mb[16 + c16];
        #pragma unroll
        for (int i = 0; i < 8; i++) {
            float a0 = s0[i] * scale + bias0;
            float a1 = s1[i] * scale + bias1;
            float tmax = fmaxf(a0, a1);
            tmax = fmaxf(tmax, __shfl_xor(tmax, 1));
            tmax = fmaxf(tmax, __shfl_xor(tmax, 2));
            tmax = fmaxf(tmax, __shfl_xor(tmax, 4));
            tmax = fmaxf(tmax, __shfl_xor(tmax, 8));   // within 16-lane half
            float newm = fmaxf(mrow[i], tmax);
            float corr = __expf(mrow[i] - newm);
            float p0 = __expf(a0 - newm);
            float p1 = __expf(a1 - newm);
            float rs = p0 + p1;
            rs += __shfl_xor(rs, 1);
            rs += __shfl_xor(rs, 2);
            rs += __shfl_xor(rs, 4);
            rs += __shfl_xor(rs, 8);
            lrow[i] = lrow[i] * corr + rs;
            mrow[i] = newm;
            #pragma unroll
            for (int j = 0; j < 4; j++) acc[j][i] = acc[j][i] * corr;
            int prow = i + hf * 8;                     // C-layout row of this VGPR
            Pw[prow * 40 + c16]      = f2bf(p0);
            Pw[prow * 40 + 16 + c16] = f2bf(p1);
        }

        // O += P(16x32) @ V(32x64): preload 4 V^T fragments, then 4 WMMAs
        v16bf pf = frag_ld(Pw, 40, 0, 0, lane);
        v16bf vf[4];
        #pragma unroll
        for (int j = 0; j < 4; j++) vf[j] = frag_ld(Vt, 40, j * 16, 0, lane);
        #pragma unroll
        for (int j = 0; j < 4; j++) acc[j] = wmma_bf16(pf, vf[j], acc[j]);
    }

    #pragma unroll
    for (int j = 0; j < 4; j++) {
        #pragma unroll
        for (int i = 0; i < 8; i++) {
            size_t row = baseRow + q0 + wave * 16 + i + hf * 8;
            out[row * EMB + hh * HD + j * 16 + c16] = acc[j][i] / lrow[i];
        }
    }
}

// ---------------------------------------------------------------------------
// Orchestration
// ---------------------------------------------------------------------------
extern "C" void kernel_launch(void* const* d_in, const int* in_sizes, int n_in,
                              void* d_out, int out_size, void* d_ws, size_t ws_size,
                              hipStream_t stream) {
    (void)in_sizes; (void)n_in; (void)out_size; (void)ws_size;
    const float*         x_in = (const float*)d_in[0];
    const unsigned char* mask = (const unsigned char*)d_in[1];
    const float* wq = (const float*)d_in[2];
    const float* bq = (const float*)d_in[3];
    const float* wk = (const float*)d_in[4];
    const float* bk = (const float*)d_in[5];
    const float* wv = (const float*)d_in[6];
    const float* bv = (const float*)d_in[7];
    const float* g1 = (const float*)d_in[8];
    const float* b1 = (const float*)d_in[9];
    const float* g2 = (const float*)d_in[10];
    const float* b2 = (const float*)d_in[11];
    const float* gf = (const float*)d_in[12];
    const float* bf = (const float*)d_in[13];
    const float* W1 = (const float*)d_in[14];
    const float* B1 = (const float*)d_in[15];
    const float* W2 = (const float*)d_in[16];
    const float* B2 = (const float*)d_in[17];

    char* p = (char*)d_ws;
    auto alloc = [&](size_t bytes) {
        void* r = (void*)p;
        p += (bytes + 255) & ~(size_t)255;
        return r;
    };
    float*          x    = (float*)alloc((size_t)ROWS * EMB * 4);
    float*          att  = (float*)alloc((size_t)ROWS * EMB * 4);
    unsigned short* hbf  = (unsigned short*)alloc((size_t)ROWS * EMB * 2);
    unsigned short* qbf  = (unsigned short*)alloc((size_t)ROWS * EMB * 2);
    unsigned short* kbf  = (unsigned short*)alloc((size_t)ROWS * EMB * 2);
    unsigned short* vbf  = (unsigned short*)alloc((size_t)ROWS * EMB * 2);
    unsigned short* ffbf = (unsigned short*)alloc((size_t)ROWS * DFF_ * 2);
    unsigned short* wqb  = (unsigned short*)alloc((size_t)NL * EMB * EMB * 2);
    unsigned short* wkb  = (unsigned short*)alloc((size_t)NL * EMB * EMB * 2);
    unsigned short* wvb  = (unsigned short*)alloc((size_t)NL * EMB * EMB * 2);
    unsigned short* W1b  = (unsigned short*)alloc((size_t)DFF_ * EMB * 2);
    unsigned short* W2b  = (unsigned short*)alloc((size_t)EMB * DFF_ * 2);

    hipMemcpyAsync(x, x_in, (size_t)ROWS * EMB * 4, hipMemcpyDeviceToDevice, stream);

    const int nW = NL * EMB * EMB;
    k_f32_to_bf16<<<(nW + 255) / 256, 256, 0, stream>>>(wq, wqb, nW);
    k_f32_to_bf16<<<(nW + 255) / 256, 256, 0, stream>>>(wk, wkb, nW);
    k_f32_to_bf16<<<(nW + 255) / 256, 256, 0, stream>>>(wv, wvb, nW);
    const int nF = DFF_ * EMB;
    k_f32_to_bf16<<<(nF + 255) / 256, 256, 0, stream>>>(W1, W1b, nF);
    k_f32_to_bf16<<<(nF + 255) / 256, 256, 0, stream>>>(W2, W2b, nF);

    for (int l = 0; l < NL; l++) {
        // h = LN(x, g1, b1)  (bf16 for WMMA)
        k_layernorm<<<ROWS, 256, 0, stream>>>(x, g1, b1, nullptr, nullptr, hbf);
        // q,k,v = h @ W^T + b  (bf16 out)
        k_gemm_bf16<<<dim3(EMB / 64, ROWS / 256), 256, 0, stream>>>(
            hbf, wqb + (size_t)l * EMB * EMB, bq + l * EMB, nullptr, nullptr, qbf,
            ROWS, EMB, EMB, 0);
        k_gemm_bf16<<<dim3(EMB / 64, ROWS / 256), 256, 0, stream>>>(
            hbf, wkb + (size_t)l * EMB * EMB, bk + l * EMB, nullptr, nullptr, kbf,
            ROWS, EMB, EMB, 0);
        k_gemm_bf16<<<dim3(EMB / 64, ROWS / 256), 256, 0, stream>>>(
            hbf, wvb + (size_t)l * EMB * EMB, bv + l * EMB, nullptr, nullptr, vbf,
            ROWS, EMB, EMB, 0);
        // flash attention
        k_attention<<<dim3(SEQ / 64, NH, BDIM), 128, 0, stream>>>(qbf, kbf, vbf, mask, att);
        // x = x + LN(att, g2, b2)
        k_layernorm<<<ROWS, 256, 0, stream>>>(att, g2, b2, x, x, nullptr);
        // d = LN_noaffine(x) (bf16)
        k_layernorm<<<ROWS, 256, 0, stream>>>(x, nullptr, nullptr, nullptr, nullptr, hbf);
        // ff = relu(d @ W1^T + B1)
        k_gemm_bf16<<<dim3(DFF_ / 64, ROWS / 256), 256, 0, stream>>>(
            hbf, W1b, B1, nullptr, nullptr, ffbf, ROWS, DFF_, EMB, 1);
        // x = x + ff @ W2^T + B2
        k_gemm_bf16<<<dim3(EMB / 64, ROWS / 256), 256, 0, stream>>>(
            ffbf, W2b, B2, x, x, nullptr, ROWS, EMB, DFF_, 0);
    }
    // out = LN(x, gf, bf)
    k_layernorm<<<ROWS, 256, 0, stream>>>(x, gf, bf, nullptr, (float*)d_out, nullptr);
}
